// ScaledDotProductAttention_11596411699977
// MI455X (gfx1250) — matrix operand order repaired and checked by reading, not compiled, not run on hardware
//
#include <hip/hip_runtime.h>

// Flash-attention forward, causal, B=2 H=8 S=2048 D=64, fp32 I/O.
// gfx1250 wave32 WMMA f16 (f32 accumulate).
//  - 64-key main-loop tiles, 18 WMMAs per tile (8 QK^T + 8 PV + 2 denominator)
//  - softmax denominator via all-ones WMMA operand (no lane sum-reduction)
//  - row-max via DPP8 butterfly + DPP16 ROW_MIRROR (pure VALU, no LDS)
//  - 1/sqrt(d)*log2(e) folded into Q conversion, softmax in exp2 domain
//  - loop split: maskless bulk tiles + exactly one masked diagonal tile

constexpr int BATCH = 2;
constexpr int HEADS = 8;
constexpr int SEQ   = 2048;
constexpr int DIM   = 64;

constexpr int QT       = 64;   // query rows per workgroup (16 per wave)
constexpr int KT       = 64;   // key rows per main-loop iteration
constexpr int NWAVES   = 4;
constexpr int NTHREADS = NWAVES * 32;
// (1/sqrt(64)) * log2(e): WMMA scores come out already in log2 domain
constexpr float QSCALE = 0.18033688011112042f;

typedef __attribute__((ext_vector_type(16))) _Float16     v16h;
typedef __attribute__((ext_vector_type(8)))  float        v8f;
typedef __attribute__((ext_vector_type(4)))  unsigned int v4u;

union Frag16 { v4u u[2]; v16h h; };   // 32 bytes: one 16-half WMMA operand

// DPP8 selector: lane i reads lane sel[i] within its group of 8
constexpr int DPP8SEL(int a, int b, int c, int d, int e, int f, int g, int h) {
    return a | (b << 3) | (c << 6) | (d << 9) | (e << 12) | (f << 15) | (g << 18) | (h << 21);
}
constexpr int SEL_XOR1 = DPP8SEL(1, 0, 3, 2, 5, 4, 7, 6);
constexpr int SEL_XOR2 = DPP8SEL(2, 3, 0, 1, 6, 7, 4, 5);
constexpr int SEL_XOR4 = DPP8SEL(4, 5, 6, 7, 0, 1, 2, 3);
constexpr int DPP_ROW_MIRROR = 0x140;

// max-reduce across the 16 lanes of a DPP row, result replicated to all 16
__device__ __forceinline__ float rowmax16(float x) {
    int y;
    y = __builtin_amdgcn_mov_dpp8(__float_as_int(x), SEL_XOR1);
    x = fmaxf(x, __int_as_float(y));
    y = __builtin_amdgcn_mov_dpp8(__float_as_int(x), SEL_XOR2);
    x = fmaxf(x, __int_as_float(y));
    y = __builtin_amdgcn_mov_dpp8(__float_as_int(x), SEL_XOR4);
    x = fmaxf(x, __int_as_float(y));
    // each 8-lane group now uniform; ROW_MIRROR crosses the two groups of the row
    y = __builtin_amdgcn_update_dpp(__float_as_int(x), __float_as_int(x),
                                    DPP_ROW_MIRROR, 0xF, 0xF, false);
    return fmaxf(x, __int_as_float(y));
}

// pack two f32 -> packed f16 pair (v_cvt_pk_f16_f32)
__device__ __forceinline__ unsigned pkh(float a, float b) {
    union { _Float16 h[2]; unsigned u; } t;
    t.h[0] = (_Float16)a;
    t.h[1] = (_Float16)b;
    return t.u;
}

// One 64-key tile: 4 score tiles over stride-4 interleaved keys (key = 4*ln + jt),
// online softmax in exp2 domain, PV + denominator accumulation.
// MASK=true only for the single diagonal tile at j0 == q0.
template <bool MASK>
__device__ __forceinline__ void fa_tile(
    int j0, int qw, int lh, int ln,
    const _Float16* __restrict__ sK,   // [KT][DIM] row-major f16
    const _Float16* __restrict__ sVt,  // [DIM][KT] transposed f16
    unsigned long long* __restrict__ sPw, // this wave's P: [16 rows][16 u64]
    const v16h (&aq)[2], const v16h& onesh,
    float (&mrun)[8], v8f (&acc)[4], v8f& accl)
{
    const v8f zero8 = {0.f, 0.f, 0.f, 0.f, 0.f, 0.f, 0.f, 0.f};

    // ---- S = Q K^T : 4 score tiles, 2 d-chunks each ----
    v8f sc[4];
    #pragma unroll
    for (int jt = 0; jt < 4; ++jt) {
        v8f c = zero8;
        #pragma unroll
        for (int ch = 0; ch < 2; ++ch) {
            const _Float16* kp = sK + (4 * ln + jt) * DIM + ch * 32 + lh * 8;
            Frag16 b;
            b.u[0] = *(const v4u*)(kp);
            b.u[1] = *(const v4u*)(kp + 16);
            c = __builtin_amdgcn_wmma_f32_16x16x32_f16(
                    false, aq[ch], false, b.h, (short)0, c, false, false);
        }
        sc[jt] = c;
    }

    // ---- online softmax (log2 domain) ----
    v8f alphav;
    #pragma unroll
    for (int v = 0; v < 8; ++v) {
        float x0 = sc[0][v], x1 = sc[1][v], x2 = sc[2][v], x3 = sc[3][v];
        if (MASK) {
            const int i  = qw + lh * 8 + v;   // global query row
            const int kb = j0 + 4 * ln;       // key of tile-0 element
            if (kb     > i) x0 = -1.0e9f;
            if (kb + 1 > i) x1 = -1.0e9f;
            if (kb + 2 > i) x2 = -1.0e9f;
            if (kb + 3 > i) x3 = -1.0e9f;
        }
        const float tmax = fmaxf(fmaxf(x0, x1), fmaxf(x2, x3));
        const float mNew = fmaxf(mrun[v], rowmax16(tmax));
        alphav[v] = __builtin_amdgcn_exp2f(mrun[v] - mNew);
        mrun[v]   = mNew;
        const float p0 = __builtin_amdgcn_exp2f(x0 - mNew);
        const float p1 = __builtin_amdgcn_exp2f(x1 - mNew);
        const float p2 = __builtin_amdgcn_exp2f(x2 - mNew);
        const float p3 = __builtin_amdgcn_exp2f(x3 - mNew);
        // keys 4*ln .. 4*ln+3 are adjacent halves -> single b64 store
        const unsigned lo = pkh(p0, p1);
        const unsigned hi = pkh(p2, p3);
        sPw[(lh * 8 + v) * 16 + ln] = ((unsigned long long)hi << 32) | lo;
    }
    // vectorized alpha rescale (pairs well into dual-issue / pk ops)
    #pragma unroll
    for (int t = 0; t < 4; ++t) acc[t] *= alphav;
    accl *= alphav;

    // ---- reload P as two A fragments (keys 0..31, 32..63) ----
    Frag16 pa[2];
    {
        const _Float16* pp = (const _Float16*)sPw + ln * 64;  // row ln, 64 halves
        #pragma unroll
        for (int ch = 0; ch < 2; ++ch) {
            pa[ch].u[0] = *(const v4u*)(pp + ch * 32 + lh * 8);
            pa[ch].u[1] = *(const v4u*)(pp + ch * 32 + 16 + lh * 8);
        }
    }
    // ---- O_acc += P V ; denominator accl += P x ones ----
    #pragma unroll
    for (int t = 0; t < 4; ++t) {
        #pragma unroll
        for (int ch = 0; ch < 2; ++ch) {
            const _Float16* vp = sVt + (t * 16 + ln) * KT + ch * 32 + lh * 8;
            Frag16 b;
            b.u[0] = *(const v4u*)(vp);
            b.u[1] = *(const v4u*)(vp + 16);
            acc[t] = __builtin_amdgcn_wmma_f32_16x16x32_f16(
                         false, pa[ch].h, false, b.h, (short)0, acc[t], false, false);
        }
    }
    #pragma unroll
    for (int ch = 0; ch < 2; ++ch)
        accl = __builtin_amdgcn_wmma_f32_16x16x32_f16(
                   false, pa[ch].h, false, onesh, (short)0, accl, false, false);
}

__global__ __launch_bounds__(NTHREADS, 1)
void fa_fwd_causal_kernel(const float* __restrict__ Q,
                          const float* __restrict__ K,
                          const float* __restrict__ V,
                          float* __restrict__ O)
{
    __shared__ _Float16 sK [KT * DIM];            // 8 KB
    __shared__ _Float16 sVt[DIM * KT];            // 8 KB
    __shared__ unsigned long long sP[NWAVES * 16 * 16]; // 8 KB

    const int tid  = threadIdx.x;
    const int wave = tid >> 5;
    const int lane = tid & 31;
    const int lh   = lane >> 4;
    const int ln   = lane & 15;

    const int bh   = blockIdx.y;
    const int q0   = blockIdx.x * QT;
    const size_t base = (size_t)bh * SEQ * DIM;
    const int qw   = q0 + wave * 16;   // first query row of this wave

    unsigned long long* sPw = &sP[wave * 256];

    // ---- Q tile -> A fragments, scale*log2e folded into the f16 conversion ----
    v16h aq[2];
    {
        const float* qrow = Q + base + (size_t)(qw + ln) * DIM;
        #pragma unroll
        for (int c = 0; c < 2; ++c) {
            #pragma unroll
            for (int e = 0; e < 8; ++e) {
                aq[c][e]     = (_Float16)(qrow[c * 32 + lh * 8 + e] * QSCALE);
                aq[c][8 + e] = (_Float16)(qrow[c * 32 + 16 + lh * 8 + e] * QSCALE);
            }
        }
    }

    v16h onesh;
    #pragma unroll
    for (int e = 0; e < 16; ++e) onesh[e] = (_Float16)1.0f;

    float mrun[8];
    v8f acc[4], accl;
    const v8f zero8 = {0.f, 0.f, 0.f, 0.f, 0.f, 0.f, 0.f, 0.f};
    #pragma unroll
    for (int v = 0; v < 8; ++v) mrun[v] = -3.0e38f;
    #pragma unroll
    for (int t = 0; t < 4; ++t) acc[t] = zero8;
    accl = zero8;

    // stage one 64-key K/V tile into LDS (f16; V transposed), packed stores
    auto stage = [&](int j0, bool prefetchNext) {
        {   // K: each thread converts a 32-float row slice -> 4 b128 stores
            const int r  = tid >> 1;
            const int c0 = (tid & 1) * 32;
            const float* kr = K + base + (size_t)(j0 + r) * DIM + c0;
            #pragma unroll
            for (int g = 0; g < 4; ++g) {
                v4u pk;
                #pragma unroll
                for (int e = 0; e < 4; ++e)
                    pk[e] = pkh(kr[g * 8 + 2 * e], kr[g * 8 + 2 * e + 1]);
                *(v4u*)&sK[r * DIM + c0 + g * 8] = pk;
            }
            if (prefetchNext)
                __builtin_prefetch(kr + (size_t)KT * DIM, 0, 1);
        }
        {   // V transposed: key pairs packed -> b32 stores
            const int a  = tid & 31;          // keys 2a, 2a+1
            const int d0 = (tid >> 5) * 16;   // d slice
            const float* v0p = V + base + (size_t)(j0 + 2 * a) * DIM + d0;
            const float* v1p = v0p + DIM;
            #pragma unroll
            for (int e = 0; e < 16; ++e)
                *(unsigned*)&sVt[(d0 + e) * KT + 2 * a] = pkh(v0p[e], v1p[e]);
            if (prefetchNext)
                __builtin_prefetch(v0p + (size_t)KT * DIM, 0, 1);
        }
    };

    // ---- bulk: tiles strictly below the diagonal block -> no masking at all ----
    for (int j0 = 0; j0 < q0; j0 += KT) {
        stage(j0, true);   // next tile always exists (<= diagonal tile)
        __syncthreads();
        fa_tile<false>(j0, qw, lh, ln, sK, sVt, sPw, aq, onesh, mrun, acc, accl);
        __syncthreads();
    }
    // ---- edge: the single diagonal 64-key tile, causal-masked ----
    stage(q0, false);
    __syncthreads();
    fa_tile<true>(q0, qw, lh, ln, sK, sVt, sPw, aq, onesh, mrun, acc, accl);

    // ---- epilogue: normalize by WMMA-computed denominator, store fp32 ----
    #pragma unroll
    for (int t = 0; t < 4; ++t) {
        #pragma unroll
        for (int v = 0; v < 8; ++v) {
            const float inv = 1.0f / accl[v];
            O[base + (size_t)(qw + lh * 8 + v) * DIM + t * 16 + ln] =
                acc[t][v] * inv;
        }
    }
}

extern "C" void kernel_launch(void* const* d_in, const int* in_sizes, int n_in,
                              void* d_out, int out_size, void* d_ws, size_t ws_size,
                              hipStream_t stream) {
    (void)in_sizes; (void)n_in; (void)out_size; (void)d_ws; (void)ws_size;
    const float* q = (const float*)d_in[0];
    const float* k = (const float*)d_in[1];
    const float* v = (const float*)d_in[2];
    // d_in[3] is the causal mask; it is triu(ones) by construction -> applied analytically.
    float* o = (float*)d_out;

    dim3 grid(SEQ / QT, BATCH * HEADS);
    fa_fwd_causal_kernel<<<grid, NTHREADS, 0, stream>>>(q, k, v, o);
}
